// MultiHeadAttentionLayer_867583394155
// MI455X (gfx1250) — matrix operand order repaired and checked by reading, not compiled, not run on hardware
//
#include <hip/hip_runtime.h>
#include <hip/hip_bf16.h>

#define DMODEL 1024
#define NHEADS 16
#define DKH    64
#define SEQ    2048
#define BATCH  2
#define MROWS  (BATCH*SEQ)

typedef __attribute__((ext_vector_type(16))) __bf16 bf16x16;
typedef __attribute__((ext_vector_type(8)))  float  f32x8;
typedef __attribute__((ext_vector_type(4)))  int    v4i;

union Frag { uint4 q[2]; bf16x16 v; };

#if __has_builtin(__builtin_amdgcn_global_load_async_to_lds_b128) && \
    __has_builtin(__builtin_amdgcn_s_wait_asynccnt)
#define HAVE_ASYNC 1
#else
#define HAVE_ASYNC 0
#endif

// 16-byte global -> LDS copy.  Async (ASYNCcnt-tracked, no VGPR round trip)
// when the toolchain exposes the gfx1250 builtin, else synchronous fallback.
// Builtin signature (from the round-2 diagnostic): the pointer params are
// v4i in address spaces 1 (global) and 3 (LDS).
__device__ __forceinline__ void cp16_g2l(const void* g, void* l) {
#if HAVE_ASYNC
  __builtin_amdgcn_global_load_async_to_lds_b128(
      (__attribute__((address_space(1))) v4i*)(g),
      (__attribute__((address_space(3))) v4i*)(l), 0, 0);
#else
  *(uint4*)l = *(const uint4*)g;
#endif
}

__device__ __forceinline__ void async_fence() {
#if HAVE_ASYNC
  __builtin_amdgcn_s_wait_asynccnt(0);
#endif
}

__device__ __forceinline__ f32x8 wmma_bf16f32(bf16x16 a, bf16x16 b, f32x8 c) {
  // 8 args: (neg_a, A, neg_b, B, c_mod, C, reuse_a, reuse_b)
  return __builtin_amdgcn_wmma_f32_16x16x32_bf16(false, a, false, b, (short)0, c,
                                                 false, false);
}

__device__ __forceinline__ unsigned pk2(float lo, float hi) {
  unsigned short a = __builtin_bit_cast(unsigned short, (__bf16)lo);
  unsigned short b = __builtin_bit_cast(unsigned short, (__bf16)hi);
  return (unsigned)a | ((unsigned)b << 16);
}

__device__ __forceinline__ uint4 cvt8(float4 a, float4 b) {
  uint4 u;
  u.x = pk2(a.x, a.y); u.y = pk2(a.z, a.w);
  u.z = pk2(b.x, b.y); u.w = pk2(b.z, b.w);
  return u;
}

// LDS tiles: R rows x 32 bf16 = 4 x uint4 per row.  Fragment pick matches the
// 16-bit WMMA A/B layout: lanes 0-15 take uint4 chunks {c, c+2} (K 0-7/16-23),
// lanes 16-31 take {c+1, c+3} (K 8-15/24-31).
//
// One K-step: each wave owns an (MT*16) x (NT*16) slab; B fragments are loaded
// once and reused across all MT A-subtiles -> MT*NT WMMAs per barrier pair.
template <int MT, int NT>
__device__ __forceinline__ void mma_tiles(const uint4 (*sA)[4], const uint4 (*sB)[4],
                                          int amBase, int bnBase, int lane,
                                          f32x8 acc[MT][NT]) {
  const int lo = lane & 15, hi = lane >> 4;
  Frag b[NT], a;
#pragma unroll
  for (int n = 0; n < NT; ++n) {
    const int bn = bnBase + n * 16 + lo;
    b[n].q[0] = sB[bn][hi];
    b[n].q[1] = sB[bn][hi + 2];
  }
#pragma unroll
  for (int m = 0; m < MT; ++m) {
    const int r = amBase + m * 16 + lo;
    a.q[0] = sA[r][hi];
    a.q[1] = sA[r][hi + 2];
#pragma unroll
    for (int n = 0; n < NT; ++n) acc[m][n] = wmma_bf16f32(a.v, b[n].v, acc[m][n]);
  }
}

// ---------------------------------------------------------------------------
// Kernel 1: QKV projection.  Y = X @ W^T + b, bf16 into [B,H,S,64].
// Workgroup tile 128x128, waves 2x4, each wave 64x32 (4x2 WMMA subtiles).
// grid: (MROWS/128, DMODEL/128, 3)  z selects Q/K/V.
// ---------------------------------------------------------------------------
__global__ __launch_bounds__(256) void qkv_proj_kernel(
    const float* __restrict__ Xq, const float* __restrict__ Xk, const float* __restrict__ Xv,
    const float* __restrict__ Wq, const float* __restrict__ Wk, const float* __restrict__ Wv,
    const float* __restrict__ bq, const float* __restrict__ bk, const float* __restrict__ bv,
    __hip_bfloat16* __restrict__ Qh, __hip_bfloat16* __restrict__ Kh,
    __hip_bfloat16* __restrict__ Vh) {
  const int z = blockIdx.z;
  const float* X = (z == 0) ? Xq : (z == 1) ? Xk : Xv;
  const float* W = (z == 0) ? Wq : (z == 1) ? Wk : Wv;
  const float* bias = (z == 0) ? bq : (z == 1) ? bk : bv;
  __hip_bfloat16* Y = (z == 0) ? Qh : (z == 1) ? Kh : Vh;

  __shared__ uint4 sA[128][4];
  __shared__ uint4 sB[128][4];

  const int t = threadIdx.x;
  const int lane = t & 31, w = t >> 5;
  const int amBase = (w >> 2) * 64, bnBase = (w & 3) * 32;
  const int mbase = blockIdx.x * 128, nbase = blockIdx.y * 128;
  const int srow = t >> 1, shalf = t & 1;  // stage: 1 row, 16 elems per thread

  f32x8 acc[4][2] = {};

  for (int kb = 0; kb < DMODEL; kb += 32) {
    const float* ap = X + (size_t)(mbase + srow) * DMODEL + kb + shalf * 16;
    const float* bp = W + (size_t)(nbase + srow) * DMODEL + kb + shalf * 16;
    if (kb + 32 < DMODEL) {
      __builtin_prefetch(ap + 32, 0, 3);
      __builtin_prefetch(bp + 32, 0, 3);
    }
    const float4* a4 = (const float4*)ap;
    const float4* b4 = (const float4*)bp;
    sA[srow][shalf * 2 + 0] = cvt8(a4[0], a4[1]);
    sA[srow][shalf * 2 + 1] = cvt8(a4[2], a4[3]);
    sB[srow][shalf * 2 + 0] = cvt8(b4[0], b4[1]);
    sB[srow][shalf * 2 + 1] = cvt8(b4[2], b4[3]);
    __syncthreads();
    mma_tiles<4, 2>(sA, sB, amBase, bnBase, lane, acc);
    __syncthreads();
  }

  const int lo = lane & 15, hi = lane >> 4;
#pragma unroll
  for (int n = 0; n < 2; ++n) {
    const int col = nbase + bnBase + n * 16 + lo;
    const int h = col >> 6, dk = col & 63;
    const float bb = bias[col];
#pragma unroll
    for (int m = 0; m < 4; ++m) {
      const int rbase = mbase + amBase + m * 16 + hi * 8;
#pragma unroll
      for (int j = 0; j < 8; ++j) {
        const int row = rbase + j;
        const int b = row >> 11, sq = row & (SEQ - 1);
        const size_t idx = ((((size_t)b * NHEADS + h) * SEQ) + sq) * DKH + dk;
        Y[idx] = __float2bfloat16(acc[m][n][j] + bb);
      }
    }
  }
}

// ---------------------------------------------------------------------------
// Kernel 2: scores = Q @ K^T * (1/sqrt(64)), fp32 into d_out attention region.
// bf16 operands staged with async global->LDS copies.
// grid: (SEQ/128, SEQ/128, B*H)
// ---------------------------------------------------------------------------
__global__ __launch_bounds__(256) void scores_kernel(
    const __hip_bfloat16* __restrict__ Qh, const __hip_bfloat16* __restrict__ Kh,
    float* __restrict__ attn) {
  const int bh = blockIdx.z;
  const __hip_bfloat16* Q = Qh + (size_t)bh * SEQ * DKH;
  const __hip_bfloat16* K = Kh + (size_t)bh * SEQ * DKH;
  float* out = attn + (size_t)bh * SEQ * SEQ;

  __shared__ uint4 sA[128][4];
  __shared__ uint4 sB[128][4];

  const int t = threadIdx.x;
  const int lane = t & 31, w = t >> 5;
  const int amBase = (w >> 2) * 64, bnBase = (w & 3) * 32;
  const int mbase = blockIdx.x * 128, nbase = blockIdx.y * 128;
  const int srow = t >> 1, shalf = t & 1;

  f32x8 acc[4][2] = {};

  for (int kb = 0; kb < DKH; kb += 32) {
    const __hip_bfloat16* qp = Q + (size_t)(mbase + srow) * DKH + kb + shalf * 16;
    const __hip_bfloat16* kp = K + (size_t)(nbase + srow) * DKH + kb + shalf * 16;
    cp16_g2l(qp + 0, &sA[srow][shalf * 2 + 0]);
    cp16_g2l(qp + 8, &sA[srow][shalf * 2 + 1]);
    cp16_g2l(kp + 0, &sB[srow][shalf * 2 + 0]);
    cp16_g2l(kp + 8, &sB[srow][shalf * 2 + 1]);
    async_fence();
    __syncthreads();
    mma_tiles<4, 2>(sA, sB, amBase, bnBase, lane, acc);
    __syncthreads();
  }

  const int lo = lane & 15, hi = lane >> 4;
#pragma unroll
  for (int n = 0; n < 2; ++n) {
    const int col = nbase + bnBase + n * 16 + lo;
#pragma unroll
    for (int m = 0; m < 4; ++m) {
      const int rbase = mbase + amBase + m * 16 + hi * 8;
#pragma unroll
      for (int j = 0; j < 8; ++j) {
        out[(size_t)(rbase + j) * SEQ + col] = acc[m][n][j] * 0.125f;
      }
    }
  }
}

// ---------------------------------------------------------------------------
// Kernel 3: row softmax over attention, in place.  One block per row.
// grid: (B*H*SEQ)
// ---------------------------------------------------------------------------
__global__ __launch_bounds__(256) void softmax_kernel(float* __restrict__ attn) {
  __shared__ float srow[SEQ];
  __shared__ float rmax[8];
  __shared__ float rsum[8];

  float* p = attn + (size_t)blockIdx.x * SEQ;
  const int t = threadIdx.x;
  const int lane = t & 31, w = t >> 5;

  float m = -3.402823466e38f;
  for (int i = t; i < SEQ; i += 256) {
    float v = p[i];
    srow[i] = v;
    m = fmaxf(m, v);
  }
  for (int off = 16; off > 0; off >>= 1) m = fmaxf(m, __shfl_xor(m, off, 32));
  if (lane == 0) rmax[w] = m;
  __syncthreads();
  const float gm = fmaxf(fmaxf(fmaxf(rmax[0], rmax[1]), fmaxf(rmax[2], rmax[3])),
                         fmaxf(fmaxf(rmax[4], rmax[5]), fmaxf(rmax[6], rmax[7])));

  float sum = 0.0f;
  for (int i = t; i < SEQ; i += 256) {
    float e = __expf(srow[i] - gm);
    srow[i] = e;
    sum += e;
  }
  for (int off = 16; off > 0; off >>= 1) sum += __shfl_xor(sum, off, 32);
  if (lane == 0) rsum[w] = sum;
  __syncthreads();
  const float gs = ((rsum[0] + rsum[1]) + (rsum[2] + rsum[3])) +
                   ((rsum[4] + rsum[5]) + (rsum[6] + rsum[7]));
  const float inv = 1.0f / gs;

  for (int i = t; i < SEQ; i += 256) p[i] = srow[i] * inv;
}

// ---------------------------------------------------------------------------
// Kernel 4: context head = attention @ V.  A fp32 (convert-on-stage), V bf16
// stored [K,N] so staged transposed.  Tile 128x64, waves 2x4, each wave 64x16.
// Output bf16 ctx [B*S, DMODEL] at col h*64.
// grid: (SEQ/128, 1, B*H)
// ---------------------------------------------------------------------------
__global__ __launch_bounds__(256) void pv_kernel(
    const float* __restrict__ attn, const __hip_bfloat16* __restrict__ Vh,
    __hip_bfloat16* __restrict__ ctx) {
  const int bh = blockIdx.z;
  const int b = bh >> 4, h = bh & 15;
  const float* P = attn + (size_t)bh * SEQ * SEQ;
  const unsigned short* V = (const unsigned short*)(Vh + (size_t)bh * SEQ * DKH);

  __shared__ uint4 sA[128][4];
  __shared__ uint4 sB[64][4];

  const int t = threadIdx.x;
  const int lane = t & 31, w = t >> 5;
  const int amBase = (w >> 2) * 64, bnBase = (w & 3) * 16;
  const int mbase = blockIdx.x * 128;
  const int srow = t >> 1, shalf = t & 1;    // A stage
  const int vn = t >> 2, vchunk = t & 3;     // B (transposed) stage

  f32x8 acc[4][1] = {};

  for (int kb = 0; kb < SEQ; kb += 32) {
    const float* ap = P + (size_t)(mbase + srow) * SEQ + kb + shalf * 16;
    if (kb + 32 < SEQ) __builtin_prefetch(ap + 32, 0, 3);
    const float4* a4 = (const float4*)ap;
    sA[srow][shalf * 2 + 0] = cvt8(a4[0], a4[1]);
    sA[srow][shalf * 2 + 1] = cvt8(a4[2], a4[3]);
    {  // sB[n][chunk] <- V[kb + chunk*8 + 0..7][n]
      const unsigned short* vp = V + (size_t)(kb + vchunk * 8) * DKH + vn;
      uint4 u;
      u.x = (unsigned)vp[0 * DKH] | ((unsigned)vp[1 * DKH] << 16);
      u.y = (unsigned)vp[2 * DKH] | ((unsigned)vp[3 * DKH] << 16);
      u.z = (unsigned)vp[4 * DKH] | ((unsigned)vp[5 * DKH] << 16);
      u.w = (unsigned)vp[6 * DKH] | ((unsigned)vp[7 * DKH] << 16);
      sB[vn][vchunk] = u;
    }
    __syncthreads();
    mma_tiles<4, 1>(sA, sB, amBase, bnBase, lane, acc);
    __syncthreads();
  }

  const int lo = lane & 15, hi = lane >> 4;
  const int coln = bnBase + lo;  // 0..63 within head
#pragma unroll
  for (int m = 0; m < 4; ++m) {
    const int rbase = mbase + amBase + m * 16 + hi * 8;
#pragma unroll
    for (int j = 0; j < 8; ++j) {
      const size_t idx = (size_t)(b * SEQ + rbase + j) * DMODEL + h * DKH + coln;
      ctx[idx] = __float2bfloat16(acc[m][0][j]);
    }
  }
}

// ---------------------------------------------------------------------------
// Kernel 5: out = ctx @ Wo^T + bo, fp32 into d_out.  ctx staged async (bf16).
// grid: (MROWS/128, DMODEL/128)
// ---------------------------------------------------------------------------
__global__ __launch_bounds__(256) void out_proj_kernel(
    const __hip_bfloat16* __restrict__ ctx, const float* __restrict__ Wo,
    const float* __restrict__ bo, float* __restrict__ out) {
  __shared__ uint4 sA[128][4];
  __shared__ uint4 sB[128][4];

  const int t = threadIdx.x;
  const int lane = t & 31, w = t >> 5;
  const int amBase = (w >> 2) * 64, bnBase = (w & 3) * 32;
  const int mbase = blockIdx.x * 128, nbase = blockIdx.y * 128;
  const int srow = t >> 1, shalf = t & 1;

  f32x8 acc[4][2] = {};

  for (int kb = 0; kb < DMODEL; kb += 32) {
    const __hip_bfloat16* cp =
        ctx + (size_t)(mbase + srow) * DMODEL + kb + shalf * 16;
    cp16_g2l(cp + 0, &sA[srow][shalf * 2 + 0]);
    cp16_g2l(cp + 8, &sA[srow][shalf * 2 + 1]);
    const float* bp = Wo + (size_t)(nbase + srow) * DMODEL + kb + shalf * 16;
    if (kb + 32 < DMODEL) __builtin_prefetch(bp + 32, 0, 3);
    const float4* b4 = (const float4*)bp;
    sB[srow][shalf * 2 + 0] = cvt8(b4[0], b4[1]);
    sB[srow][shalf * 2 + 1] = cvt8(b4[2], b4[3]);
    async_fence();
    __syncthreads();
    mma_tiles<4, 2>(sA, sB, amBase, bnBase, lane, acc);
    __syncthreads();
  }

  const int lo = lane & 15, hi = lane >> 4;
#pragma unroll
  for (int n = 0; n < 2; ++n) {
    const int col = nbase + bnBase + n * 16 + lo;
    const float bb = bo[col];
#pragma unroll
    for (int m = 0; m < 4; ++m) {
      const int rbase = mbase + amBase + m * 16 + hi * 8;
#pragma unroll
      for (int j = 0; j < 8; ++j) {
        out[(size_t)(rbase + j) * DMODEL + col] = acc[m][n][j] + bb;
      }
    }
  }
}

// ---------------------------------------------------------------------------

extern "C" void kernel_launch(void* const* d_in, const int* in_sizes, int n_in,
                              void* d_out, int out_size, void* d_ws, size_t ws_size,
                              hipStream_t stream) {
  const float* q  = (const float*)d_in[0];
  const float* k  = (const float*)d_in[1];
  const float* v  = (const float*)d_in[2];
  const float* Wq = (const float*)d_in[3];
  const float* bq = (const float*)d_in[4];
  const float* Wk = (const float*)d_in[5];
  const float* bk = (const float*)d_in[6];
  const float* Wv = (const float*)d_in[7];
  const float* bv = (const float*)d_in[8];
  const float* Wo = (const float*)d_in[9];
  const float* bo = (const float*)d_in[10];

  float* out  = (float*)d_out;
  float* attn = out + (size_t)MROWS * DMODEL;  // [B,H,S,S] fp32

  // Workspace: Qh | Kh | Vh | ctx, each MROWS*DMODEL bf16 (8 MB) = 32 MB.
  __hip_bfloat16* Qh  = (__hip_bfloat16*)d_ws;
  __hip_bfloat16* Kh  = Qh + (size_t)MROWS * DMODEL;
  __hip_bfloat16* Vh  = Kh + (size_t)MROWS * DMODEL;
  __hip_bfloat16* ctx = Vh + (size_t)MROWS * DMODEL;

  dim3 blk(256);
  qkv_proj_kernel<<<dim3(MROWS / 128, DMODEL / 128, 3), blk, 0, stream>>>(
      q, k, v, Wq, Wk, Wv, bq, bk, bv, Qh, Kh, Vh);
  scores_kernel<<<dim3(SEQ / 128, SEQ / 128, BATCH * NHEADS), blk, 0, stream>>>(
      Qh, Kh, attn);
  softmax_kernel<<<dim3(BATCH * NHEADS * SEQ), blk, 0, stream>>>(attn);
  pv_kernel<<<dim3(SEQ / 128, 1, BATCH * NHEADS), blk, 0, stream>>>(attn, Vh, ctx);
  out_proj_kernel<<<dim3(MROWS / 128, DMODEL / 128, 1), blk, 0, stream>>>(
      ctx, Wo, bo, out);
}
